// InternalMAFE_59270548684863
// MI455X (gfx1250) — matrix operand order repaired and checked by reading, not compiled, not run on hardware
//
#include <hip/hip_runtime.h>
#include <hip/hip_bf16.h>
#include <math.h>

typedef __attribute__((ext_vector_type(16))) __bf16 v16bf;
typedef __attribute__((ext_vector_type(2)))  __bf16 v2bf;
typedef __attribute__((ext_vector_type(8)))  float  v8f;

#define BMT 128
#define BNT 128
#define BKT 32
#define SM_CH 16

// Generic C = scale * (A @ B) + bias, bf16 WMMA with f32 accumulate.
// A element (m,k) at A[m*aRow + k*aCol]  (aCol=7 absorbs the slice gather)
// B element (k,n) at Bm[k*bK + n*bN]     (bK==1 => k-contiguous, transposed access)
// Requires: M % 128 == 0, Kdim % 32 == 0. N handled with guards.
// 8 waves as 4(M) x 2(N); each wave computes 32x64 = 2x4 WMMA tiles.
__global__ __launch_bounds__(256)
void gemm_bf16_wmma(const float* __restrict__ A, long aRow, long aCol,
                    const float* __restrict__ Bm, long bK, long bN,
                    float* __restrict__ C, long cRow,
                    int M, int N, int Kdim, float scale,
                    const float* __restrict__ bias)
{
    (void)M;
    // Fragment-major LDS: each lane's 16 bf16 operand contiguous (32B) ->
    // fragment read = two ds_load_b128, zero shuffling.
    __shared__ __align__(32) __bf16 sAf[8][32][16];   // eight 16x32 A-subtiles
    __shared__ __align__(32) __bf16 sBf[8][32][16];   // eight 32x16 B-subtiles

    const int tid  = threadIdx.x;
    const int lane = tid & 31;
    const int wave = tid >> 5;
    const int wm   = wave >> 1;            // 0..3  (M direction, 32 rows each)
    const int wn   = wave & 1;             // 0..1  (N direction, 64 cols each)
    const int m0   = blockIdx.y * BMT;
    const int n0   = blockIdx.x * BNT;

    const bool trB = (bK == 1);

    // ---- hoisted per-thread staging OFFSETS (keep base pointers global) ----
    // A: 128x32 tile = 2048 k-pairs, 8 per thread.
    size_t  offA[8];
    __bf16* sAp[8];
    #pragma unroll
    for (int t = 0; t < 8; ++t) {
        int p = tid + t * 256;
        int m = p >> 4;                    // 16 k-pairs per row
        int k = (p & 15) << 1;
        offA[t] = (size_t)(m0 + m) * aRow + (size_t)k * aCol;
        int fl = (m & 15) + ((k >= 16) ? 16 : 0);
        int fe = k & 15;
        sAp[t] = &sAf[m >> 4][fl][fe];
    }
    const size_t stepA = (size_t)BKT * aCol;

    // B: 32x128 tile = 2048 pairs, 8 per thread (pairing axis depends on layout).
    size_t  offB0[8], offB1[8];
    __bf16* sBp0[8];
    __bf16* sBp1[8];
    bool    bv0[8], bv1[8];
    #pragma unroll
    for (int t = 0; t < 8; ++t) {
        int p = tid + t * 256;
        if (trB) {                         // pair along k (k contiguous in memory)
            int n = p >> 4;
            int k = (p & 15) << 1;
            int gn = n0 + n;
            bv0[t] = bv1[t] = (gn < N);
            int gnc = bv0[t] ? gn : (N - 1);            // clamp: always in-bounds
            offB0[t] = (size_t)gnc * bN + (size_t)k;
            offB1[t] = offB0[t] + 1;
            int fl = (n & 15) + ((k >= 16) ? 16 : 0);
            sBp0[t] = &sBf[n >> 4][fl][k & 15];
            sBp1[t] = sBp0[t] + 1;
        } else {                           // pair along n (n contiguous in memory)
            int k = p >> 6;                // 64 n-pairs per k row
            int n = (p & 63) << 1;
            int gn = n0 + n;
            bv0[t] = (gn < N);
            bv1[t] = (gn + 1 < N);
            int gnc0 = bv0[t] ? gn       : (N - 1);
            int gnc1 = bv1[t] ? (gn + 1) : (N - 1);
            offB0[t] = (size_t)k * bK + gnc0;
            offB1[t] = (size_t)k * bK + gnc1;
            int lb = (k >= 16) ? 16 : 0;
            int fe = k & 15;
            sBp0[t] = &sBf[n >> 4][(n & 15) + lb][fe];
            sBp1[t] = &sBf[n >> 4][((n + 1) & 15) + lb][fe];
        }
    }
    const size_t stepB = trB ? (size_t)BKT : (size_t)BKT * bK;

    v8f acc[2][4] = {};

    const int ksteps = Kdim / BKT;
    for (int ks = 0; ks < ksteps; ++ks) {
        // ---- stage A (global_load_b32 pairs -> cvt_pk_bf16 -> ds_store_b32) ----
        #pragma unroll
        for (int t = 0; t < 8; ++t) {
            float v0 = A[offA[t]];
            float v1 = A[offA[t] + (size_t)aCol];
            v2bf pv; pv.x = (__bf16)v0; pv.y = (__bf16)v1;
            *(v2bf*)sAp[t] = pv;
            offA[t] += stepA;
        }
        // ---- stage B ----
        if (trB) {
            #pragma unroll
            for (int t = 0; t < 8; ++t) {
                float v0 = Bm[offB0[t]];
                float v1 = Bm[offB1[t]];
                if (!bv0[t]) { v0 = 0.0f; v1 = 0.0f; }
                v2bf pv; pv.x = (__bf16)v0; pv.y = (__bf16)v1;
                *(v2bf*)sBp0[t] = pv;
                offB0[t] += stepB; offB1[t] += stepB;
            }
        } else {
            #pragma unroll
            for (int t = 0; t < 8; ++t) {
                float v0 = Bm[offB0[t]];
                float v1 = Bm[offB1[t]];
                if (!bv0[t]) v0 = 0.0f;
                if (!bv1[t]) v1 = 0.0f;
                *sBp0[t] = (__bf16)v0;
                *sBp1[t] = (__bf16)v1;
                offB0[t] += stepB; offB1[t] += stepB;
            }
        }
        __syncthreads();

        // prefetch next K-step source lines
        if (ks + 1 < ksteps) {
            __builtin_prefetch(&A[offA[0]], 0, 3);
            __builtin_prefetch(&Bm[offB0[0]], 0, 3);
        }

        // ---- fragment loads: contiguous 32B per lane ----
        v16bf af[2], bf2[4];
        #pragma unroll
        for (int tm = 0; tm < 2; ++tm)
            af[tm] = *(const v16bf*)&sAf[wm * 2 + tm][lane][0];
        #pragma unroll
        for (int tn = 0; tn < 4; ++tn)
            bf2[tn] = *(const v16bf*)&sBf[wn * 4 + tn][lane][0];

        // ---- 8 WMMA per wave per K-step ----
        #pragma unroll
        for (int tm = 0; tm < 2; ++tm)
            #pragma unroll
            for (int tn = 0; tn < 4; ++tn)
                acc[tm][tn] = __builtin_amdgcn_wmma_f32_16x16x32_bf16(
                    false, af[tm], false, bf2[tn],
                    (short)0, acc[tm][tn], false, false);
        __syncthreads();
    }

    // ---- store C (C/D layout: VGPR r -> M = r + (lane<16 ? 0 : 8)) ----
    #pragma unroll
    for (int tm = 0; tm < 2; ++tm) {
        #pragma unroll
        for (int tn = 0; tn < 4; ++tn) {
            int col = n0 + wn * 64 + tn * 16 + (lane & 15);
            if (col < N) {
                int rbase = m0 + wm * 32 + tm * 16 + ((lane < 16) ? 0 : 8);
                float bv = bias ? bias[col] : 0.0f;
                #pragma unroll
                for (int r = 0; r < 8; ++r)
                    C[(size_t)(rbase + r) * cRow + col] = acc[tm][tn][r] * scale + bv;
            }
        }
    }
}

// Online softmax over the BATCH axis, coalesced: thread j owns column j and
// streams a contiguous chunk of rows. blockDim.x == Ncols (288 = 9 waves).
__global__ void softmax_partial(const float* __restrict__ ST,
                                float* __restrict__ pm, float* __restrict__ ps,
                                int Bsz, int Ncols)
{
    const int i = blockIdx.x, ch = blockIdx.y, j = threadIdx.x;
    const int rows = Bsz / SM_CH;
    const int b0 = ch * rows, b1 = b0 + rows;
    const float* base = ST + (size_t)i * Bsz * Ncols;
    float m = -INFINITY, s = 0.0f;
    for (int b = b0; b < b1; ++b) {
        float v = base[(size_t)b * Ncols + j];
        float mn = fmaxf(m, v);
        s = s * __expf(m - mn) + __expf(v - mn);
        m = mn;
    }
    pm[(size_t)(i * SM_CH + ch) * Ncols + j] = m;
    ps[(size_t)(i * SM_CH + ch) * Ncols + j] = s;
}

__global__ void softmax_combine(const float* __restrict__ pm,
                                const float* __restrict__ ps,
                                float* __restrict__ stats, int Ncols)
{
    const int i = blockIdx.x, j = threadIdx.x;
    float m = -INFINITY;
    for (int c = 0; c < SM_CH; ++c)
        m = fmaxf(m, pm[(size_t)(i * SM_CH + c) * Ncols + j]);
    float s = 0.0f;
    for (int c = 0; c < SM_CH; ++c)
        s += ps[(size_t)(i * SM_CH + c) * Ncols + j] *
             __expf(pm[(size_t)(i * SM_CH + c) * Ncols + j] - m);
    stats[(i * Ncols + j) * 2 + 0] = m;
    stats[(i * Ncols + j) * 2 + 1] = 1.0f / s;
}

// o_i = V_i * softmax(ST_i), then the 7-step gated scan, interleaved output.
__global__ __launch_bounds__(256)
void scan_weight_kernel(const float* __restrict__ ST, const float* __restrict__ V,
                        const float* __restrict__ stats,
                        const float* __restrict__ pa1, const float* __restrict__ pa2,
                        const float* __restrict__ pb1, const float* __restrict__ pb2,
                        float* __restrict__ o1, int Bsz, int Ncols, int P)
{
    int idx = blockIdx.x * 256 + threadIdx.x;
    if (idx >= Bsz * Ncols) return;
    int b = idx / Ncols, j = idx - b * Ncols;
    float a1 = pa1[0], a2 = pa2[0], b1 = pb1[0], b2 = pb2[0];
    float y = 0.0f;
    for (int i = 0; i < P; ++i) {
        size_t off = ((size_t)i * Bsz + b) * Ncols + j;
        float mxv = stats[(i * Ncols + j) * 2 + 0];
        float rsv = stats[(i * Ncols + j) * 2 + 1];
        float att = __expf(ST[off] - mxv) * rsv;
        float o = V[off] * att;
        if (i == 0) {
            y = o;
        } else {
            float t = tanhf(a1 * y + b1);
            float s = 1.0f / (1.0f + __expf(-(a2 * y + b2)));
            y = t * s + o;
        }
        o1[(size_t)b * (Ncols * P) + (size_t)j * P + i] = y;
    }
}

extern "C" void kernel_launch(void* const* d_in, const int* in_sizes, int n_in,
                              void* d_out, int out_size, void* d_ws, size_t ws_size,
                              hipStream_t stream)
{
    (void)in_sizes; (void)n_in; (void)out_size; (void)ws_size;
    const float* x   = (const float*)d_in[0];
    const float* wk1 = (const float*)d_in[1];
    const float* wv1 = (const float*)d_in[3];
    const float* h1  = (const float*)d_in[5];
    const float* a1  = (const float*)d_in[7];
    const float* a2  = (const float*)d_in[8];
    const float* b1  = (const float*)d_in[9];
    const float* b2  = (const float*)d_in[10];
    const float* plw = (const float*)d_in[13];
    const float* plb = (const float*)d_in[14];
    float* out = (float*)d_out;
    // Branch 2 (w_k2/w_v2/h2) and proj_w/proj_b feed only dead code in the
    // jitted reference (DCE'd) — intentionally not computed.

    const int Bsz = 4096, INP = 2016, SEQ = 1024, N1 = 288, P = 7;
    const size_t slice = (size_t)Bsz * N1;

    float* Kbuf  = (float*)d_ws;                     // [7][B][N1]
    float* Vbuf  = Kbuf  + (size_t)P * slice;        // [7][B][N1]
    float* STbuf = Vbuf  + (size_t)P * slice;        // [7][B][N1]
    float* stats = STbuf + (size_t)P * slice;        // [7][N1][2]
    float* pm    = stats + (size_t)P * N1 * 2;       // [7][SM_CH][N1]
    float* ps    = pm    + (size_t)P * SM_CH * N1;   // [7][SM_CH][N1]
    float* o1    = ps    + (size_t)P * SM_CH * N1;   // [B][2016]

    dim3 blk(256);
    dim3 gKV((N1 + BNT - 1) / BNT, Bsz / BMT);       // (3, 32)
    const float sc = 1.0f / sqrtf((float)N1);

    for (int i = 0; i < P; ++i) {
        gemm_bf16_wmma<<<gKV, blk, 0, stream>>>(x + i, (long)INP, (long)P,
            wk1, (long)N1, 1L, Kbuf + (size_t)i * slice, (long)N1,
            Bsz, N1, N1, 1.0f, nullptr);
        gemm_bf16_wmma<<<gKV, blk, 0, stream>>>(x + i, (long)INP, (long)P,
            wv1, (long)N1, 1L, Vbuf + (size_t)i * slice, (long)N1,
            Bsz, N1, N1, 1.0f, nullptr);
    }
    for (int i = 0; i < P; ++i) {
        gemm_bf16_wmma<<<gKV, blk, 0, stream>>>(Kbuf + (size_t)i * slice, (long)N1, 1L,
            h1, 1L, (long)N1, STbuf + (size_t)i * slice, (long)N1,
            Bsz, N1, N1, sc, nullptr);
    }
    softmax_partial<<<dim3(P, SM_CH), dim3(N1), 0, stream>>>(STbuf, pm, ps, Bsz, N1);
    softmax_combine<<<dim3(P), dim3(N1), 0, stream>>>(pm, ps, stats, N1);
    scan_weight_kernel<<<dim3((Bsz * N1 + 255) / 256), blk, 0, stream>>>(
        STbuf, Vbuf, stats, a1, a2, b1, b2, o1, Bsz, N1, P);
    // out = o1 @ proj_len_w.T + proj_len_b   (M=4096, N=1024, K=2016)
    gemm_bf16_wmma<<<dim3(SEQ / BNT, Bsz / BMT), blk, 0, stream>>>(o1, (long)INP, 1L,
        plw, 1L, (long)INP, out, (long)SEQ, Bsz, SEQ, INP, 1.0f, plb);
}
